// CustomGNN_63058709840500
// MI455X (gfx1250) — compile-verified, matrix operand order checked by reading, not run on hardware
//
#include <hip/hip_runtime.h>

typedef __attribute__((ext_vector_type(16))) __bf16 v16bf;
typedef __attribute__((ext_vector_type(8)))  float  v8f;

#define N_NODES 100000
#define N_EDGES 1250000
#define D_IN    64
#define D_HID   128
#define D_OUT   3
#define WAVES_PER_BLOCK 4

// ---------------------------------------------------------------------------
// Kernel 1: zero the f32 aggregation workspace (must run every launch).
// ---------------------------------------------------------------------------
__global__ void gnn_zero_kernel(float4* __restrict__ p, int n4) {
    int i = blockIdx.x * blockDim.x + threadIdx.x;
    if (i < n4) p[i] = make_float4(0.f, 0.f, 0.f, 0.f);
}

// ---------------------------------------------------------------------------
// Kernel 2: edge scatter-add.  thread = (edge, 16-byte chunk of D_IN).
// aggr[dst] += feat[src] * w  via f32 atomics.
// ---------------------------------------------------------------------------
__global__ void gnn_edge_scatter_kernel(const float* __restrict__ feat,
                                        const long long* __restrict__ edges,
                                        const float* __restrict__ ew,
                                        float* __restrict__ aggr) {
    const int CHUNKS = D_IN / 4;  // 16 chunks of float4
    long long t = (long long)blockIdx.x * blockDim.x + threadIdx.x;
    long long e = t / CHUNKS;
    int c = (int)(t % CHUNKS);
    if (e >= N_EDGES) return;
    int src = (int)edges[e];
    int dst = (int)edges[N_EDGES + e];
    float w = ew[e];
    const float4 f = *(const float4*)(feat + (long long)src * D_IN + c * 4);
    float* a = aggr + (long long)dst * D_IN + c * 4;
    atomicAdd(a + 0, f.x * w);
    atomicAdd(a + 1, f.y * w);
    atomicAdd(a + 2, f.z * w);
    atomicAdd(a + 3, f.w * w);
}

// ---------------------------------------------------------------------------
// WMMA fragment helpers.  16-bit A-matrix 16x32 layout (ISA 7.12.2):
//   lane half = lane>>4, m = lane&15
//   VGPR v (pair j=2v,2v+1):  k = (v>=4)*16 + half*8 + (v&3)*2 + (j&1)
// B 32x16 mirrors A with n = lane&15 selecting the column, B[k][n] = W[n][k].
// C/D f32: vgpr r -> row m = r + half*8, col n = lane&15.
// ---------------------------------------------------------------------------
__device__ __forceinline__ int frag_k(int v, int half) {
    return ((v & 4) ? 16 : 0) + half * 8 + (v & 3) * 2;
}

// A fragment, layer 1: columns 0..63 from x, 64..127 from aggr (concat).
__device__ __forceinline__ v16bf load_A_concat(const float* __restrict__ x,
                                               const float* __restrict__ aggr,
                                               int node, int lane, int kchunk) {
    int half = lane >> 4;
    v16bf a;
#pragma unroll
    for (int v = 0; v < 8; ++v) {
        int k = kchunk * 32 + frag_k(v, half);
        const float* srcp = (k < D_IN) ? (x + (long long)node * D_IN + k)
                                       : (aggr + (long long)node * D_IN + (k - D_IN));
        a[2 * v]     = (__bf16)srcp[0];
        a[2 * v + 1] = (__bf16)srcp[1];
    }
    return a;
}

// A fragment from LDS hidden buffer (row-major [16 nodes][128] bf16).
__device__ __forceinline__ v16bf load_A_lds(const __bf16* __restrict__ h,
                                            int lane, int kchunk) {
    int half = lane >> 4, m = lane & 15;
    v16bf a;
#pragma unroll
    for (int v = 0; v < 8; ++v) {
        int k = kchunk * 32 + frag_k(v, half);
        a[2 * v]     = h[m * D_HID + k];
        a[2 * v + 1] = h[m * D_HID + k + 1];
    }
    return a;
}

// B fragment from LDS weights (row-major [out][128] bf16), column tile nt.
__device__ __forceinline__ v16bf load_B_lds(const __bf16* __restrict__ w,
                                            int nt, int lane, int kchunk) {
    int half = lane >> 4, n = lane & 15;
    v16bf b;
#pragma unroll
    for (int v = 0; v < 8; ++v) {
        int k = kchunk * 32 + frag_k(v, half);
        b[2 * v]     = w[(nt * 16 + n) * D_HID + k];
        b[2 * v + 1] = w[(nt * 16 + n) * D_HID + k + 1];
    }
    return b;
}

// ---------------------------------------------------------------------------
// Kernel 3: fused 3-layer MLP.  4 waves/block, one 16-node tile per wave.
// ---------------------------------------------------------------------------
__global__ void __launch_bounds__(32 * WAVES_PER_BLOCK)
gnn_mlp_kernel(const float* __restrict__ x, const float* __restrict__ aggr,
               const float* __restrict__ W_rel, const float* __restrict__ b_rel,
               const float* __restrict__ W_h1, const float* __restrict__ b_h1,
               const float* __restrict__ W_out, const float* __restrict__ b_out,
               float* __restrict__ out, int n_nodes) {
    __shared__ __bf16 sW[D_HID * D_HID];                       // 32 KB, restaged per layer
    __shared__ __bf16 sH[WAVES_PER_BLOCK][16 * D_HID];         // 4 x 4 KB hidden tiles

    const int tid = threadIdx.x;
    const int nthr = blockDim.x;
    const int wave = tid >> 5;
    const int lane = tid & 31;
    const int half = lane >> 4;
    const int m = lane & 15;
    const int n = lane & 15;

    const int tile = blockIdx.x * WAVES_PER_BLOCK + wave;
    const int n_tiles = (n_nodes + 15) / 16;
    const bool valid = tile < n_tiles;
    const int node_base = (valid ? tile : 0) * 16;

    __bf16* myH = sH[wave];

    // ---- Layer 1: h = relu(concat(x, aggr) @ W_rel^T + b_rel) --------------
    for (int i = tid; i < D_HID * D_HID; i += nthr) sW[i] = (__bf16)W_rel[i];
    v16bf a0 = load_A_concat(x, aggr, node_base + m, lane, 0);
    v16bf a1 = load_A_concat(x, aggr, node_base + m, lane, 1);
    v16bf a2 = load_A_concat(x, aggr, node_base + m, lane, 2);
    v16bf a3 = load_A_concat(x, aggr, node_base + m, lane, 3);
    __syncthreads();
#pragma unroll
    for (int nt = 0; nt < D_HID / 16; ++nt) {
        float bias = b_rel[nt * 16 + n];
        v8f acc;
#pragma unroll
        for (int r = 0; r < 8; ++r) acc[r] = bias;
        acc = __builtin_amdgcn_wmma_f32_16x16x32_bf16(false, a0, false, load_B_lds(sW, nt, lane, 0), (short)0, acc, false, false);
        acc = __builtin_amdgcn_wmma_f32_16x16x32_bf16(false, a1, false, load_B_lds(sW, nt, lane, 1), (short)0, acc, false, false);
        acc = __builtin_amdgcn_wmma_f32_16x16x32_bf16(false, a2, false, load_B_lds(sW, nt, lane, 2), (short)0, acc, false, false);
        acc = __builtin_amdgcn_wmma_f32_16x16x32_bf16(false, a3, false, load_B_lds(sW, nt, lane, 3), (short)0, acc, false, false);
#pragma unroll
        for (int r = 0; r < 8; ++r)
            myH[(r + half * 8) * D_HID + nt * 16 + n] = (__bf16)fmaxf(acc[r], 0.f);
    }
    __syncthreads();

    // ---- Layer 2: h = relu(h @ W_h1^T + b_h1) ------------------------------
    for (int i = tid; i < D_HID * D_HID; i += nthr) sW[i] = (__bf16)W_h1[i];
    a0 = load_A_lds(myH, lane, 0);
    a1 = load_A_lds(myH, lane, 1);
    a2 = load_A_lds(myH, lane, 2);
    a3 = load_A_lds(myH, lane, 3);
    __syncthreads();
#pragma unroll
    for (int nt = 0; nt < D_HID / 16; ++nt) {
        float bias = b_h1[nt * 16 + n];
        v8f acc;
#pragma unroll
        for (int r = 0; r < 8; ++r) acc[r] = bias;
        acc = __builtin_amdgcn_wmma_f32_16x16x32_bf16(false, a0, false, load_B_lds(sW, nt, lane, 0), (short)0, acc, false, false);
        acc = __builtin_amdgcn_wmma_f32_16x16x32_bf16(false, a1, false, load_B_lds(sW, nt, lane, 1), (short)0, acc, false, false);
        acc = __builtin_amdgcn_wmma_f32_16x16x32_bf16(false, a2, false, load_B_lds(sW, nt, lane, 2), (short)0, acc, false, false);
        acc = __builtin_amdgcn_wmma_f32_16x16x32_bf16(false, a3, false, load_B_lds(sW, nt, lane, 3), (short)0, acc, false, false);
#pragma unroll
        for (int r = 0; r < 8; ++r)
            myH[(r + half * 8) * D_HID + nt * 16 + n] = (__bf16)fmaxf(acc[r], 0.f);
    }
    __syncthreads();

    // ---- Layer 3: out = h @ W_out^T + b_out  (3 outputs, one 16-wide tile) -
    for (int i = tid; i < 16 * D_HID; i += nthr)
        sW[i] = (i < D_OUT * D_HID) ? (__bf16)W_out[i] : (__bf16)0.f;
    a0 = load_A_lds(myH, lane, 0);
    a1 = load_A_lds(myH, lane, 1);
    a2 = load_A_lds(myH, lane, 2);
    a3 = load_A_lds(myH, lane, 3);
    __syncthreads();
    {
        float bias = (n < D_OUT) ? b_out[n] : 0.f;
        v8f acc;
#pragma unroll
        for (int r = 0; r < 8; ++r) acc[r] = bias;
        acc = __builtin_amdgcn_wmma_f32_16x16x32_bf16(false, a0, false, load_B_lds(sW, 0, lane, 0), (short)0, acc, false, false);
        acc = __builtin_amdgcn_wmma_f32_16x16x32_bf16(false, a1, false, load_B_lds(sW, 0, lane, 1), (short)0, acc, false, false);
        acc = __builtin_amdgcn_wmma_f32_16x16x32_bf16(false, a2, false, load_B_lds(sW, 0, lane, 2), (short)0, acc, false, false);
        acc = __builtin_amdgcn_wmma_f32_16x16x32_bf16(false, a3, false, load_B_lds(sW, 0, lane, 3), (short)0, acc, false, false);
        if (valid && n < D_OUT) {
#pragma unroll
            for (int r = 0; r < 8; ++r) {
                int node = node_base + half * 8 + r;
                if (node < n_nodes) out[(long long)node * D_OUT + n] = acc[r];
            }
        }
    }
}

// ---------------------------------------------------------------------------
// Host launcher
// ---------------------------------------------------------------------------
extern "C" void kernel_launch(void* const* d_in, const int* in_sizes, int n_in,
                              void* d_out, int out_size, void* d_ws, size_t ws_size,
                              hipStream_t stream) {
    const float*     feat   = (const float*)d_in[0];
    const long long* edges  = (const long long*)d_in[1];  // int64 [2, E]
    const float*     ew     = (const float*)d_in[2];
    const float*     W_rel  = (const float*)d_in[3];
    const float*     b_rel  = (const float*)d_in[4];
    const float*     W_h1   = (const float*)d_in[5];
    const float*     b_h1   = (const float*)d_in[6];
    const float*     W_out  = (const float*)d_in[7];
    const float*     b_out  = (const float*)d_in[8];
    float*           outp   = (float*)d_out;
    float*           aggr   = (float*)d_ws;               // N_NODES * D_IN f32

    // 1) zero aggregation buffer
    {
        int n4 = (N_NODES * D_IN) / 4;
        int blocks = (n4 + 255) / 256;
        gnn_zero_kernel<<<blocks, 256, 0, stream>>>((float4*)aggr, n4);
    }
    // 2) weighted scatter-add over edges
    {
        long long threads = (long long)N_EDGES * (D_IN / 4);
        int blocks = (int)((threads + 255) / 256);
        gnn_edge_scatter_kernel<<<blocks, 256, 0, stream>>>(feat, edges, ew, aggr);
    }
    // 3) fused 3-layer MLP with bf16 WMMA
    {
        int n_tiles = (N_NODES + 15) / 16;                       // 6250
        int blocks = (n_tiles + WAVES_PER_BLOCK - 1) / WAVES_PER_BLOCK;
        gnn_mlp_kernel<<<blocks, 32 * WAVES_PER_BLOCK, 0, stream>>>(
            feat, aggr, W_rel, b_rel, W_h1, b_h1, W_out, b_out, outp, N_NODES);
    }
}
